// GobyBlock_6571299963260
// MI455X (gfx1250) — matrix operand order repaired and checked by reading, not compiled, not run on hardware
//
#include <hip/hip_runtime.h>
#include <cstddef>

// ---------------- CDNA5 WMMA types ----------------
typedef __attribute__((ext_vector_type(16))) __bf16 v16bf;
typedef __attribute__((ext_vector_type(8)))  float  v8f;
typedef int v4i_vs __attribute__((vector_size(4 * sizeof(int))));  // matches builtin param

#define AS1 __attribute__((address_space(1)))
#define AS3 __attribute__((address_space(3)))

#if defined(__gfx1250__) && __has_builtin(__builtin_amdgcn_global_load_async_to_lds_b128) && __has_builtin(__builtin_amdgcn_s_wait_asynccnt)
#define HAVE_ASYNC_LDS 1
#else
#define HAVE_ASYNC_LDS 0
#endif

union Frag {                 // one 16x32 (A) or 32x16 (B) bf16 fragment per lane
    v16bf  v;
    float4 f4[2];
};

__device__ inline v8f wmma_bf16(const Frag& a, const Frag& b, v8f c) {
    return __builtin_amdgcn_wmma_f32_16x16x32_bf16(
        /*neg_a=*/false, a.v, /*neg_b=*/false, b.v,
        /*c_mod=*/(short)0, c, /*reuse_a=*/false, /*reuse_b=*/false);
}

#if HAVE_ASYNC_LDS
// 16-byte global -> LDS async copy (per-lane addresses), tracked by ASYNCcnt
__device__ inline void async_copy16(const __bf16* g, __bf16* l) {
    __builtin_amdgcn_global_load_async_to_lds_b128(
        (AS1 v4i_vs*)(g), (AS3 v4i_vs*)(l), 0, 0);
}
#endif

// ---------------- model constants ----------------
#define MB   2
#define MT   2048
#define MD   2048
#define MNH  16
#define MNKV 4
#define MHD  128
#define MFFN 8192

// =====================================================================
// fp32 -> bf16 conversion (weights)
// =====================================================================
__global__ void f32_to_bf16_kernel(const float* __restrict__ in,
                                   __bf16* __restrict__ out, size_t n) {
    size_t i = (size_t)blockIdx.x * 256 + threadIdx.x;
    if (i < n) out[i] = (__bf16)in[i];
}

// =====================================================================
// RMSNorm: x [rows, D] fp32 -> h [rows, D] bf16
// =====================================================================
__global__ __launch_bounds__(256)
void rmsnorm_kernel(const float* __restrict__ x, const float* __restrict__ w,
                    __bf16* __restrict__ h, int D) {
    __shared__ float red[256];
    const int row = blockIdx.x;
    const float* xr = x + (size_t)row * D;
    float s = 0.f;
    for (int j = threadIdx.x; j < D; j += 256) { float v = xr[j]; s += v * v; }
    red[threadIdx.x] = s;
    __syncthreads();
    for (int st = 128; st > 0; st >>= 1) {
        if (threadIdx.x < st) red[threadIdx.x] += red[threadIdx.x + st];
        __syncthreads();
    }
    const float norm = rsqrtf(red[0] / (float)D + 1e-5f);
    __bf16* hr = h + (size_t)row * D;
    for (int j = threadIdx.x; j < D; j += 256)
        hr[j] = (__bf16)(xr[j] * norm * w[j]);
}

// =====================================================================
// Generic WMMA GEMM:  C[M,N] (+)= A[M,K] * W[N,K]^T   (A,W bf16)
//   MODE 0: C is bf16, overwrite
//   MODE 1: C is fp32, accumulate (+=)
// Block = 256 threads (8 waves). Tile 128x128, BK=32.
// Wave (w>>1, w&1) owns a 32x64 tile = 2x4 WMMA accumulators.
// Double-buffered async global->LDS staging (ASYNCcnt) when available.
// =====================================================================
template <int MODE>
__global__ __launch_bounds__(256)
void gemm_bf16_wmma(const __bf16* __restrict__ A, const __bf16* __restrict__ W,
                    void* __restrict__ Cout, int M, int N, int K) {
    __shared__ __align__(16) __bf16 As[2][128][32];
    __shared__ __align__(16) __bf16 Ws[2][128][32];

    const int tid  = threadIdx.x;
    const int lane = tid & 31;
    const int wv   = tid >> 5;
    const int nl   = lane & 15;   // column / row-in-tile selector
    const int hl   = lane >> 4;   // half-wave selector
    const int wm   = (wv >> 1) * 32;
    const int wn   = (wv & 1) * 64;
    const int rowbase = blockIdx.y * 128;
    const int colbase = blockIdx.x * 128;

    const v8f zero8 = {0.f, 0.f, 0.f, 0.f, 0.f, 0.f, 0.f, 0.f};
    v8f acc[2][4];
#pragma unroll
    for (int i = 0; i < 2; ++i)
#pragma unroll
        for (int j = 0; j < 4; ++j) acc[i][j] = zero8;

    const int lrow = tid >> 1;          // 0..127
    const int lk   = (tid & 1) * 16;    // 0 or 16
    const __bf16* gA = A + (size_t)(rowbase + lrow) * K + lk;
    const __bf16* gW = W + (size_t)(colbase + lrow) * K + lk;
    const int nk = K >> 5;

#if HAVE_ASYNC_LDS
    // ---------- async double-buffered pipeline ----------
    // 4 async b128 ops per thread per tile; loads complete in order, so
    // s_wait_asynccnt 4 == "previous tile landed" while next is in flight.
    {
        async_copy16(gA,     &As[0][lrow][lk]);
        async_copy16(gA + 8, &As[0][lrow][lk + 8]);
        async_copy16(gW,     &Ws[0][lrow][lk]);
        async_copy16(gW + 8, &Ws[0][lrow][lk + 8]);
    }
    for (int kt = 0; kt < nk; ++kt) {
        const int cur = kt & 1;
        if (kt + 1 < nk) {
            const __bf16* na = gA + (size_t)(kt + 1) * 32;
            const __bf16* nw = gW + (size_t)(kt + 1) * 32;
            async_copy16(na,     &As[cur ^ 1][lrow][lk]);
            async_copy16(na + 8, &As[cur ^ 1][lrow][lk + 8]);
            async_copy16(nw,     &Ws[cur ^ 1][lrow][lk]);
            async_copy16(nw + 8, &Ws[cur ^ 1][lrow][lk + 8]);
            __builtin_amdgcn_s_wait_asynccnt(4);
        } else {
            __builtin_amdgcn_s_wait_asynccnt(0);
        }
        __syncthreads();   // tile `cur` visible to all waves

        Frag a[2], b[4];
#pragma unroll
        for (int mt = 0; mt < 2; ++mt) {
            const __bf16* base = &As[cur][wm + mt * 16 + nl][0];
            a[mt].f4[0] = *(const float4*)(base + hl * 8);
            a[mt].f4[1] = *(const float4*)(base + 16 + hl * 8);
        }
#pragma unroll
        for (int g = 0; g < 4; ++g) {
            const __bf16* base = &Ws[cur][wn + g * 16 + nl][0];
            b[g].f4[0] = *(const float4*)(base + hl * 8);
            b[g].f4[1] = *(const float4*)(base + 16 + hl * 8);
        }
#pragma unroll
        for (int mt = 0; mt < 2; ++mt)
#pragma unroll
            for (int g = 0; g < 4; ++g)
                acc[mt][g] = wmma_bf16(a[mt], b[g], acc[mt][g]);

        __syncthreads();   // everyone done reading `cur` before it is reused
    }
#else
    // ---------- synchronous fallback ----------
    for (int kt = 0; kt < nk; ++kt) {
        const int k0 = kt * 32;
        __syncthreads();
        {
            const float4* ga = (const float4*)(gA + k0);
            float4* sa = (float4*)(&As[0][lrow][lk]);
            sa[0] = ga[0]; sa[1] = ga[1];
            const float4* gw = (const float4*)(gW + k0);
            float4* sw = (float4*)(&Ws[0][lrow][lk]);
            sw[0] = gw[0]; sw[1] = gw[1];
            if (kt + 1 < nk) {  // prefetch next K tile -> global_prefetch_b8
                __builtin_prefetch(gA + k0 + 32, 0, 0);
                __builtin_prefetch(gW + k0 + 32, 0, 0);
            }
        }
        __syncthreads();

        Frag a[2], b[4];
#pragma unroll
        for (int mt = 0; mt < 2; ++mt) {
            const __bf16* base = &As[0][wm + mt * 16 + nl][0];
            a[mt].f4[0] = *(const float4*)(base + hl * 8);
            a[mt].f4[1] = *(const float4*)(base + 16 + hl * 8);
        }
#pragma unroll
        for (int g = 0; g < 4; ++g) {
            const __bf16* base = &Ws[0][wn + g * 16 + nl][0];
            b[g].f4[0] = *(const float4*)(base + hl * 8);
            b[g].f4[1] = *(const float4*)(base + 16 + hl * 8);
        }
#pragma unroll
        for (int mt = 0; mt < 2; ++mt)
#pragma unroll
            for (int g = 0; g < 4; ++g)
                acc[mt][g] = wmma_bf16(a[mt], b[g], acc[mt][g]);
    }
#endif

    // ---- epilogue: C layout M = r + 8*hl, N = nl ----
#pragma unroll
    for (int mt = 0; mt < 2; ++mt)
#pragma unroll
        for (int g = 0; g < 4; ++g)
#pragma unroll
            for (int r = 0; r < 8; ++r) {
                const int row = rowbase + wm + mt * 16 + r + 8 * hl;
                const int col = colbase + wn + g * 16 + nl;
                const float v = acc[mt][g][r];
                if (MODE == 0)
                    ((__bf16*)Cout)[(size_t)row * N + col] = (__bf16)v;
                else
                    ((float*)Cout)[(size_t)row * N + col] += v;
            }
}

// =====================================================================
// RoPE: lin [B*T, nh*128] bf16 -> out [B, nh, T, 128] bf16
// grid = B*nh*T, block = 64 (one lane per rotation pair)
// =====================================================================
__global__ __launch_bounds__(64)
void rope_kernel(const __bf16* __restrict__ lin, const float* __restrict__ cosr,
                 const float* __restrict__ sinr, __bf16* __restrict__ out,
                 int nh, int T) {
    const int d   = threadIdx.x;            // 0..63
    const int blk = blockIdx.x;
    const int t   = blk % T;
    const int h   = (blk / T) % nh;
    const int b   = blk / (T * nh);
    const int rowlen = nh * MHD;
    const __bf16* src = lin + ((size_t)(b * T + t)) * rowlen + h * MHD;
    const float x1 = (float)src[d];
    const float x2 = (float)src[64 + d];
    const float c = cosr[t * 64 + d];
    const float s = sinr[t * 64 + d];
    __bf16* dst = out + (((size_t)(b * nh + h)) * T + t) * MHD;
    dst[d]      = (__bf16)(x1 * c - x2 * s);
    dst[64 + d] = (__bf16)(x2 * c + x1 * s);
}

// V relayout: [B*T, nkv*128] -> [B, nkv, T, 128]
__global__ __launch_bounds__(128)
void reshape_v_kernel(const __bf16* __restrict__ lin, __bf16* __restrict__ out,
                      int nh, int T) {
    const int d   = threadIdx.x;
    const int blk = blockIdx.x;
    const int t   = blk % T;
    const int h   = (blk / T) % nh;
    const int b   = blk / (T * nh);
    out[(((size_t)(b * nh + h)) * T + t) * MHD + d] =
        lin[((size_t)(b * T + t)) * (nh * MHD) + h * MHD + d];
}

// =====================================================================
// Flash attention (causal, GQA).  grid = B*NH*(T/128), block = 256.
// Each wave owns 16 query rows; 32-key tiles staged in LDS.
// S = Q*K^T via WMMA, online softmax, P*V via WMMA.
// =====================================================================
__global__ __launch_bounds__(256)
void attn_kernel(const __bf16* __restrict__ Q, const __bf16* __restrict__ Kg,
                 const __bf16* __restrict__ Vg, __bf16* __restrict__ Ctx) {
    __shared__ __align__(16) __bf16 Ks[32][128];      // [key][dim]
    __shared__ __align__(16) __bf16 Vt[128][32];      // [dim][key] (transposed)
    __shared__ __align__(16) __bf16 Pw[8][16][32];    // per-wave P scratch

    const int tid  = threadIdx.x;
    const int lane = tid & 31;
    const int wv   = tid >> 5;
    const int nl   = lane & 15;
    const int hl   = lane >> 4;

    const int blk   = blockIdx.x;
    const int qb    = (blk & 15) * 128;       // T/128 = 16 tiles
    const int head  = (blk >> 4) & 15;        // NH = 16
    const int b     = blk >> 8;
    const int kvh   = head >> 2;              // NREP = 4, repeat_interleave

    // ---- load Q fragments for this wave's 16 rows (A layout) ----
    const __bf16* qrow = Q + (((size_t)(b * MNH + head)) * MT + qb + wv * 16 + nl) * MHD;
    Frag qf[4];
#pragma unroll
    for (int c = 0; c < 4; ++c) {
        qf[c].f4[0] = *(const float4*)(qrow + c * 32 + hl * 8);
        qf[c].f4[1] = *(const float4*)(qrow + c * 32 + 16 + hl * 8);
    }

    const v8f zero8 = {0.f, 0.f, 0.f, 0.f, 0.f, 0.f, 0.f, 0.f};
    v8f cacc[8];
#pragma unroll
    for (int gd = 0; gd < 8; ++gd) cacc[gd] = zero8;
    float run_m[8], run_l[8];
#pragma unroll
    for (int r = 0; r < 8; ++r) { run_m[r] = -1e30f; run_l[r] = 0.f; }

    const size_t kvbase = ((size_t)(b * MNKV + kvh)) * MT * MHD;
    const int nkt = qb / 32 + 4;              // key tiles needed (causal)
    const float scale = 0.08838834764831845f; // 1/sqrt(128)

    for (int kt = 0; kt < nkt; ++kt) {
        __syncthreads();
        // ---- cooperative stage of K/V tile (32 keys x 128 dims) ----
        {
            const int key = tid >> 3;
            const int db  = (tid & 7) * 16;
            const size_t goff = kvbase + (size_t)(kt * 32 + key) * MHD + db;
            const float4* gk = (const float4*)(Kg + goff);
            float4* sk = (float4*)(&Ks[key][db]);
            sk[0] = gk[0]; sk[1] = gk[1];
            Frag vf;
            const float4* gv = (const float4*)(Vg + goff);
            vf.f4[0] = gv[0]; vf.f4[1] = gv[1];
#pragma unroll
            for (int i = 0; i < 16; ++i) Vt[db + i][key] = vf.v[i];
        }
        __syncthreads();

        // ---- S = Q * K^T : two 16-key groups, 4 K-dim chunks each ----
        v8f sacc[2] = {zero8, zero8};
#pragma unroll
        for (int g = 0; g < 2; ++g)
#pragma unroll
            for (int c = 0; c < 4; ++c) {
                Frag bf_;
                const __bf16* base = &Ks[g * 16 + nl][c * 32 + hl * 8];
                bf_.f4[0] = ((const float4*)base)[0];
                bf_.f4[1] = ((const float4*)base)[2];
                sacc[g] = wmma_bf16(qf[c], bf_, sacc[g]);
            }

        // ---- online softmax (row = r + 8*hl, cols across 16-lane half) ----
        float pv[2][8];
#pragma unroll
        for (int r = 0; r < 8; ++r) {
            const int q_idx = qb + wv * 16 + r + 8 * hl;
            float m_t = -1e30f;
#pragma unroll
            for (int g = 0; g < 2; ++g) {
                const int k_idx = kt * 32 + g * 16 + nl;
                float val = sacc[g][r] * scale;
                if (k_idx > q_idx) val = -1e30f;   // causal mask
                pv[g][r] = val;
                m_t = fmaxf(m_t, val);
            }
            for (int off = 8; off > 0; off >>= 1)
                m_t = fmaxf(m_t, __shfl_xor(m_t, off, 16));
            const float nm = fmaxf(run_m[r], m_t);
            const float f  = __expf(run_m[r] - nm);
            float s_t = 0.f;
#pragma unroll
            for (int g = 0; g < 2; ++g) {
                const float e = __expf(pv[g][r] - nm);
                pv[g][r] = e;
                s_t += e;
            }
            for (int off = 8; off > 0; off >>= 1)
                s_t += __shfl_xor(s_t, off, 16);
            run_l[r] = run_l[r] * f + s_t;
            run_m[r] = nm;
#pragma unroll
            for (int gd = 0; gd < 8; ++gd) cacc[gd][r] *= f;
        }

        // ---- C-layout -> A-layout via per-wave LDS scratch ----
#pragma unroll
        for (int r = 0; r < 8; ++r)
#pragma unroll
            for (int g = 0; g < 2; ++g)
                Pw[wv][r + 8 * hl][g * 16 + nl] = (__bf16)pv[g][r];
        asm volatile("s_wait_dscnt 0" ::: "memory");  // wave-internal LDS RAW

        Frag pf;
        {
            const __bf16* pbase = &Pw[wv][nl][0];
            pf.f4[0] = *(const float4*)(pbase + hl * 8);
            pf.f4[1] = *(const float4*)(pbase + 16 + hl * 8);
        }

        // ---- ctx += P * V (V transposed in LDS -> contiguous B frags) ----
#pragma unroll
        for (int gd = 0; gd < 8; ++gd) {
            Frag bv;
            const __bf16* vb = &Vt[gd * 16 + nl][hl * 8];
            bv.f4[0] = ((const float4*)vb)[0];
            bv.f4[1] = ((const float4*)vb)[2];
            cacc[gd] = wmma_bf16(pf, bv, cacc[gd]);
        }
    }

    // ---- epilogue: normalize, write ctx as [B, T, NH*HD] bf16 ----
#pragma unroll
    for (int r = 0; r < 8; ++r) {
        const float inv = 1.0f / run_l[r];
        const int row = qb + wv * 16 + r + 8 * hl;
        const size_t orow = ((size_t)b * MT + row) * (MNH * MHD) + head * MHD;
#pragma unroll
        for (int gd = 0; gd < 8; ++gd)
            Ctx[orow + gd * 16 + nl] = (__bf16)(cacc[gd][r] * inv);
    }
}

// =====================================================================
// elementwise: out = x (seed residual), silu(gate)*up
// =====================================================================
__global__ void copy_f32_kernel(const float* __restrict__ in,
                                float* __restrict__ out, size_t n) {
    size_t i = (size_t)blockIdx.x * 256 + threadIdx.x;
    if (i < n) out[i] = in[i];
}

__global__ void silu_mul_kernel(const __bf16* __restrict__ g,
                                const __bf16* __restrict__ u,
                                __bf16* __restrict__ o, size_t n) {
    size_t i = (size_t)blockIdx.x * 256 + threadIdx.x;
    if (i < n) {
        const float gf = (float)g[i];
        const float uf = (float)u[i];
        const float s = gf / (1.f + __expf(-gf));
        o[i] = (__bf16)(s * uf);
    }
}

// =====================================================================
// host-side launch
// =====================================================================
static inline size_t align_up(size_t v, size_t a) { return (v + a - 1) / a * a; }

extern "C" void kernel_launch(void* const* d_in, const int* in_sizes, int n_in,
                              void* d_out, int out_size, void* d_ws, size_t ws_size,
                              hipStream_t stream) {
    (void)in_sizes; (void)n_in; (void)out_size; (void)ws_size;
    const float* x      = (const float*)d_in[0];
    const float* cosr   = (const float*)d_in[1];
    const float* sinr   = (const float*)d_in[2];
    /* d_in[3] = mask (causal, recomputed in-kernel) */
    const float* norm_w = (const float*)d_in[4];
    const float* wq = (const float*)d_in[5];
    const float* wk = (const float*)d_in[6];
    const float* wv = (const float*)d_in[7];
    const float* wo = (const float*)d_in[8];
    const float* wg = (const float*)d_in[9];
    const float* wu = (const float*)d_in[10];
    const float* wd = (const float*)d_in[11];
    float* out = (float*)d_out;

    const size_t BT = (size_t)MB * MT;              // 4096 rows
    char* base = (char*)d_ws;
    size_t off = 0;
    auto alloc = [&](size_t bytes) -> void* {
        void* r = base + off;
        off = align_up(off + bytes, 256);
        return r;
    };
    __bf16* h_bf   = (__bf16*)alloc(BT * MD * 2);
    __bf16* wq_bf  = (__bf16*)alloc((size_t)MD * MD * 2);
    __bf16* wk_bf  = (__bf16*)alloc((size_t)MNKV * MHD * MD * 2);
    __bf16* wv_bf  = (__bf16*)alloc((size_t)MNKV * MHD * MD * 2);
    __bf16* wo_bf  = (__bf16*)alloc((size_t)MD * MD * 2);
    __bf16* wg_bf  = (__bf16*)alloc((size_t)MFFN * MD * 2);
    __bf16* wu_bf  = (__bf16*)alloc((size_t)MFFN * MD * 2);
    __bf16* wd_bf  = (__bf16*)alloc((size_t)MD * MFFN * 2);
    __bf16* q_lin  = (__bf16*)alloc(BT * MD * 2);
    __bf16* k_lin  = (__bf16*)alloc(BT * MNKV * MHD * 2);
    __bf16* v_lin  = (__bf16*)alloc(BT * MNKV * MHD * 2);
    __bf16* q_bf   = (__bf16*)alloc(BT * MD * 2);
    __bf16* k_bf   = (__bf16*)alloc(BT * MNKV * MHD * 2);
    __bf16* v_bf   = (__bf16*)alloc(BT * MNKV * MHD * 2);
    __bf16* ctx_bf = (__bf16*)alloc(BT * MD * 2);
    __bf16* gate   = (__bf16*)alloc(BT * MFFN * 2);
    __bf16* up     = (__bf16*)alloc(BT * MFFN * 2);

    auto cvt = [&](const float* src, __bf16* dst, size_t n) {
        f32_to_bf16_kernel<<<(unsigned)((n + 255) / 256), 256, 0, stream>>>(src, dst, n);
    };
    // 1) weights -> bf16
    cvt(wq, wq_bf, (size_t)MD * MD);
    cvt(wk, wk_bf, (size_t)MNKV * MHD * MD);
    cvt(wv, wv_bf, (size_t)MNKV * MHD * MD);
    cvt(wo, wo_bf, (size_t)MD * MD);
    cvt(wg, wg_bf, (size_t)MFFN * MD);
    cvt(wu, wu_bf, (size_t)MFFN * MD);
    cvt(wd, wd_bf, (size_t)MD * MFFN);

    // 2) RMSNorm
    rmsnorm_kernel<<<(unsigned)BT, 256, 0, stream>>>(x, norm_w, h_bf, MD);

    // 3) Q/K/V projections (bf16 WMMA GEMM)
    gemm_bf16_wmma<0><<<dim3(MD / 128, (unsigned)(BT / 128)), 256, 0, stream>>>(
        h_bf, wq_bf, q_lin, (int)BT, MD, MD);
    gemm_bf16_wmma<0><<<dim3((MNKV * MHD) / 128, (unsigned)(BT / 128)), 256, 0, stream>>>(
        h_bf, wk_bf, k_lin, (int)BT, MNKV * MHD, MD);
    gemm_bf16_wmma<0><<<dim3((MNKV * MHD) / 128, (unsigned)(BT / 128)), 256, 0, stream>>>(
        h_bf, wv_bf, v_lin, (int)BT, MNKV * MHD, MD);

    // 4) RoPE + GQA relayout
    rope_kernel<<<(unsigned)(MB * MNH * MT), 64, 0, stream>>>(q_lin, cosr, sinr, q_bf, MNH, MT);
    rope_kernel<<<(unsigned)(MB * MNKV * MT), 64, 0, stream>>>(k_lin, cosr, sinr, k_bf, MNKV, MT);
    reshape_v_kernel<<<(unsigned)(MB * MNKV * MT), 128, 0, stream>>>(v_lin, v_bf, MNKV, MT);

    // 5) seed residual: out = x
    copy_f32_kernel<<<(unsigned)((BT * MD + 255) / 256), 256, 0, stream>>>(x, out, BT * MD);

    // 6) flash attention
    attn_kernel<<<(unsigned)(MB * MNH * (MT / 128)), 256, 0, stream>>>(q_bf, k_bf, v_bf, ctx_bf);

    // 7) out += ctx @ wo^T
    gemm_bf16_wmma<1><<<dim3(MD / 128, (unsigned)(BT / 128)), 256, 0, stream>>>(
        ctx_bf, wo_bf, out, (int)BT, MD, MD);

    // 8) SwiGLU FFN
    gemm_bf16_wmma<0><<<dim3(MFFN / 128, (unsigned)(BT / 128)), 256, 0, stream>>>(
        h_bf, wg_bf, gate, (int)BT, MFFN, MD);
    gemm_bf16_wmma<0><<<dim3(MFFN / 128, (unsigned)(BT / 128)), 256, 0, stream>>>(
        h_bf, wu_bf, up, (int)BT, MFFN, MD);
    silu_mul_kernel<<<(unsigned)((BT * MFFN + 255) / 256), 256, 0, stream>>>(
        gate, up, gate, BT * MFFN);
    // 9) out += act @ wd^T
    gemm_bf16_wmma<1><<<dim3(MD / 128, (unsigned)(BT / 128)), 256, 0, stream>>>(
        gate, wd_bf, out, (int)BT, MD, MFFN);
}
